// PairwiseCosineSimilarity_68917045232240
// MI455X (gfx1250) — compile-verified, matrix operand order checked by reading, not run on hardware
//
#include <hip/hip_runtime.h>
#include <math.h>

// ---------------------------------------------------------------------------
// PairwiseCosineSimilarity on MI455X (gfx1250):
//   phase 1: row-normalize x1, x2 into bf16 (32 MB workspace, L2-resident)
//   phase 2: NT GEMM via v_wmma_f32_16x16x32_bf16, f32 accumulate
//
// Round-4 fixes:
//   * loop-carried per-fragment byte offsets (dedicated VGPRs, += 64/iter)
//     so address updates never alias WMMA A/B sources -> no 4-slot
//     WMMA->VALU hazard v_nops
//   * epilogue back to size_t/generic-pointer arithmetic (u32 epilogue
//     offsets defeated immediate folding and bloated valu/wait counts)
// ---------------------------------------------------------------------------

typedef float  v8f   __attribute__((ext_vector_type(8)));
typedef float  v4f   __attribute__((ext_vector_type(4)));
typedef __bf16 v16bf __attribute__((ext_vector_type(16)));
typedef __bf16 v8bf  __attribute__((ext_vector_type(8)));
typedef __bf16 v4bf  __attribute__((ext_vector_type(4)));

typedef const char  __attribute__((address_space(1)))* gbyte_cp;
typedef const v8bf  __attribute__((address_space(1)))* gv8bf_cp;

namespace {
constexpr int   K_DIM  = 1024;   // feature dim
constexpr int   N_ROWS = 8192;   // rows of x1 (output rows)
constexpr int   M_ROWS = 8192;   // rows of x2 (output cols)
constexpr float EPS    = 1e-8f;  // matches torch cosine_similarity eps

constexpr int   BM = 256;        // block tile rows (4 waves x 64)
constexpr int   BN = 128;        // block tile cols (2 waves x 64)
}

// -------------------------------------------------------------------------
// Phase 1: one block (256 threads) per row. Each thread owns 4 contiguous
// floats (256*4 == 1024). Block-reduce sum of squares, scale, emit bf16.
// Input is read exactly once -> non-temporal loads.
// -------------------------------------------------------------------------
__global__ __launch_bounds__(256) void cosim_normalize_bf16(
    const float* __restrict__ x, __bf16* __restrict__ y)
{
  const int    row  = blockIdx.x;
  const int    tid  = threadIdx.x;
  const size_t base = (size_t)row * K_DIM + (size_t)tid * 4;

  const v4f v  = __builtin_nontemporal_load((const v4f*)(x + base));
  float     ss = v[0] * v[0] + v[1] * v[1] + v[2] * v[2] + v[3] * v[3];

  __shared__ float red[256];
  red[tid] = ss;
  __syncthreads();
#pragma unroll
  for (int s = 128; s > 0; s >>= 1) {
    if (tid < s) red[tid] += red[tid + s];
    __syncthreads();
  }
  __shared__ float s_inv;
  if (tid == 0) s_inv = 1.0f / fmaxf(sqrtf(red[0]), EPS);
  __syncthreads();
  const float inv = s_inv;

  v4bf o;
  o[0] = (__bf16)(v[0] * inv);
  o[1] = (__bf16)(v[1] * inv);
  o[2] = (__bf16)(v[2] * inv);
  o[3] = (__bf16)(v[3] * inv);
  *(v4bf*)(y + base) = o;
}

// -------------------------------------------------------------------------
// Phase 2: WMMA GEMM, C = A * B^T (both operands row-major bf16 [rows, K]).
//
// Fragment layout (CDNA5 ISA 7.12.2, 16-bit A-matrix 16x32):
//   lanes 0-15  (row = lane):    VGPR0-3 = K 0..7,   VGPR4-7 = K 16..23
//   lanes 16-31 (row = lane-16): VGPR0-3 = K 8..15,  VGPR4-7 = K 24..31
// -> per lane: two 16B global loads at byte offsets (half*16) and (+32).
// NT GEMM: the B (KxN) fragment with lane = column = row of x2 has the
// identical per-lane layout, so one loader serves both operands.
// -------------------------------------------------------------------------
union FragU {
  v16bf v;
  v8bf  h[2];
};

// SADDR-form load: uniform AS(1) byte base + divergent 32-bit byte offset.
__device__ __forceinline__ v16bf load_frag_nt(gbyte_cp gbase, unsigned byteOff)
{
  FragU u;
  u.h[0] = *(gv8bf_cp)(gbase + byteOff);        // K  0..7  / K  8..15
  u.h[1] = *(gv8bf_cp)(gbase + byteOff + 32);   // K 16..23 / K 24..31
  return u.v;
}

__global__ __launch_bounds__(256) void cosim_wmma_gemm(
    const __bf16* __restrict__ A,    // [N_ROWS, K] normalized bf16
    const __bf16* __restrict__ B,    // [M_ROWS, K] normalized bf16
    float* __restrict__ C)           // [N_ROWS, M_ROWS]
{
  const int lane = threadIdx.x & 31;
  const int wid  = threadIdx.x >> 5;   // 8 waves per block
  const int wm   = wid & 3;            // 4 waves along output rows
  const int wn   = wid >> 2;           // 2 waves along output cols

  const int rowBase = blockIdx.y * BM + wm * 64;   // wave: 4 x 16 row tiles
  const int colBase = blockIdx.x * BN + wn * 64;   // wave: 4 x 16 col tiles

  const int half = lane >> 4;          // lane half selects K swizzle
  const int lrow = lane & 15;          // matrix row/col within tile

  gbyte_cp gA = (gbyte_cp)A;
  gbyte_cp gB = (gbyte_cp)B;

  // Loop-carried 32-bit BYTE offsets, one dedicated register per fragment.
  unsigned aoff[4], boff[4];
#pragma unroll
  for (int i = 0; i < 4; ++i)
    aoff[i] = ((unsigned)(rowBase + i * 16 + lrow) * K_DIM + half * 8) * 2u;
#pragma unroll
  for (int j = 0; j < 4; ++j)
    boff[j] = ((unsigned)(colBase + j * 16 + lrow) * K_DIM + half * 8) * 2u;

  v8f acc[4][4] = {};

#pragma unroll 1
  for (int ks = 0; ks < K_DIM / 32; ++ks) {       // 32 k-elems = 64 bytes/iter
    v16bf af[4];
    v16bf bfr[4];
#pragma unroll
    for (int i = 0; i < 4; ++i) af[i]  = load_frag_nt(gA, aoff[i]);
#pragma unroll
    for (int j = 0; j < 4; ++j) bfr[j] = load_frag_nt(gB, boff[j]);

#pragma unroll
    for (int i = 0; i < 4; ++i) aoff[i] += 64;    // advance while loads fly
#pragma unroll
    for (int j = 0; j < 4; ++j) boff[j] += 64;

#pragma unroll
    for (int i = 0; i < 4; ++i)
#pragma unroll
      for (int j = 0; j < 4; ++j)
        acc[i][j] = __builtin_amdgcn_wmma_f32_16x16x32_bf16(
            /*neg_a=*/false, af[i], /*neg_b=*/false, bfr[j],
            /*c_mod=*/(short)0, acc[i][j],
            /*reuse_a=*/false, /*reuse_b=*/false);
  }

  // C/D layout: VGPR r holds M = r + 8*half; lane(lrow) holds N = lrow.
  // Output is write-once and larger than L2 -> non-temporal stores.
#pragma unroll
  for (int i = 0; i < 4; ++i) {
    const int row0 = rowBase + i * 16 + half * 8;
#pragma unroll
    for (int j = 0; j < 4; ++j) {
      float* cp = C + (size_t)row0 * M_ROWS + (size_t)(colBase + j * 16 + lrow);
#pragma unroll
      for (int r = 0; r < 8; ++r)
        __builtin_nontemporal_store(acc[i][j][r], cp + (size_t)r * M_ROWS);
    }
  }
}

// -------------------------------------------------------------------------
extern "C" void kernel_launch(void* const* d_in, const int* in_sizes, int n_in,
                              void* d_out, int out_size, void* d_ws, size_t ws_size,
                              hipStream_t stream)
{
  (void)in_sizes; (void)n_in; (void)out_size; (void)ws_size;

  const float* x1  = (const float*)d_in[0];
  const float* x2  = (const float*)d_in[1];
  float*       out = (float*)d_out;

  __bf16* x1n = (__bf16*)d_ws;                       // 16 MB
  __bf16* x2n = x1n + (size_t)N_ROWS * K_DIM;        // 16 MB

  cosim_normalize_bf16<<<N_ROWS, 256, 0, stream>>>(x1, x1n);
  cosim_normalize_bf16<<<M_ROWS, 256, 0, stream>>>(x2, x2n);

  dim3 grid(M_ROWS / BN, N_ROWS / BM);               // 64 x 32 blocks
  cosim_wmma_gemm<<<grid, 256, 0, stream>>>(x1n, x2n, out);
}
